// GlobalSAModule_21809843929147
// MI455X (gfx1250) — compile-verified, matrix operand order checked by reading, not run on hardware
//
#include <hip/hip_runtime.h>
#include <math.h>

typedef __attribute__((ext_vector_type(16))) _Float16 v16h;
typedef __attribute__((ext_vector_type(8)))  _Float16 v8h;
typedef __attribute__((ext_vector_type(8)))  float    v8f;

#define NPTS   65536
#define NB_ATT 256     // attention blocks (256 rows each)

__device__ __forceinline__ float siluf(float x) { return x / (1.0f + __expf(-x)); }
__device__ __forceinline__ float sigmf(float x) { return 1.0f / (1.0f + __expf(-x)); }

// order-preserving float<->uint mapping for integer atomicMax on floats
__device__ __forceinline__ unsigned f2ord(float f) {
    unsigned u = __float_as_uint(f);
    return (u & 0x80000000u) ? ~u : (u | 0x80000000u);
}
__device__ __forceinline__ float ord2f(unsigned u) {
    unsigned v = (u & 0x80000000u) ? (u & 0x7FFFFFFFu) : ~u;
    return __uint_as_float(v);
}

// Cubic B-spline bases on uniform grid: knots g(j) = -1 + (j-3)*0.4, j=0..11.
// Produces 8 bases (GRID_SIZE + SPLINE_ORDER) matching the efficient-kan formulation.
__device__ __forceinline__ void bspline8(float x, float* out) {
    float b[11];
#pragma unroll
    for (int j = 0; j < 11; j++) {
        float gj = -1.0f + (float)(j - 3) * 0.4f;
        b[j] = (x >= gj && x < gj + 0.4f) ? 1.0f : 0.0f;
    }
#pragma unroll
    for (int j = 0; j < 10; j++) {  // k=1, denom 0.4
        float gj = -1.0f + (float)(j - 3) * 0.4f;
        b[j] = (x - gj) * 2.5f * b[j] + ((gj + 0.8f) - x) * 2.5f * b[j + 1];
    }
#pragma unroll
    for (int j = 0; j < 9; j++) {   // k=2, denom 0.8
        float gj = -1.0f + (float)(j - 3) * 0.4f;
        b[j] = (x - gj) * 1.25f * b[j] + ((gj + 1.2f) - x) * 1.25f * b[j + 1];
    }
#pragma unroll
    for (int j = 0; j < 8; j++) {   // k=3, denom 1.2
        float gj = -1.0f + (float)(j - 3) * 0.4f;
        out[j] = (x - gj) * (1.0f / 1.2f) * b[j] + ((gj + 1.6f) - x) * (1.0f / 1.2f) * b[j + 1];
    }
}

// Pack augmented weight: Wp[o][c] = c<IN ? base_w[o][c] : spline_w[o][i][t]*scaler[o][i]
template <int IN, int OUT>
__global__ void pack_w(const float* __restrict__ base_w, const float* __restrict__ spline_w,
                       const float* __restrict__ scaler, _Float16* __restrict__ Wp) {
    const int K = IN * 9;
    int idx = blockIdx.x * 256 + threadIdx.x;
    if (idx >= OUT * K) return;
    int o = idx / K, c = idx % K;
    float v;
    if (c < IN) {
        v = base_w[o * IN + c];
    } else {
        int s = c - IN, i = s >> 3;
        v = spline_w[(o * IN + i) * 8 + (s & 7)] * scaler[o * IN + i];
    }
    Wp[idx] = (_Float16)v;
}

// Fused KAN layer GEMM: Y[N,OUT] = [silu(X) | bspline(X)] @ Wp^T  (f16 WMMA, f32 accum)
// Block: 256 threads (8 waves), 64-row M tile. Input columns processed in chunks of 32:
// each chunk contributes 32 (silu) + 256 (spline) = 9 K-steps of 32.
template <int IN, int OUT>
__global__ __launch_bounds__(256) void kan_gemm(const float* __restrict__ X,
                                                const _Float16* __restrict__ Wp,
                                                float* __restrict__ Y) {
    constexpr int K   = IN * 9;
    constexpr int NCH = IN / 32;
    constexpr int NT  = OUT / 128;   // 16-col n-tiles per wave (8 waves cover OUT)
    __shared__ _Float16 sA[64][304]; // 64 x (288 + 16 pad) halves = 38.9 KB

    const int tid      = threadIdx.x;
    const int lane     = tid & 31;
    const int wave     = tid >> 5;
    const int ln       = lane & 15;
    const int halfsel  = (lane & 16) ? 8 : 0;
    const int blockRow = blockIdx.x * 64;

    v8f acc[4][NT];
#pragma unroll
    for (int mt = 0; mt < 4; mt++)
#pragma unroll
        for (int nt = 0; nt < NT; nt++)
#pragma unroll
            for (int r = 0; r < 8; r++) acc[mt][nt][r] = 0.0f;

    union HV { v16h v; v8h h[2]; };

    for (int ch = 0; ch < NCH; ch++) {
        if (ch) __syncthreads();
        const int c0 = ch * 32;
        // Phase 1: build A chunk tile in LDS (cols: [0,32)=silu, [32,288)=splines)
        for (int idx = tid; idx < 64 * 32; idx += 256) {
            int r = idx >> 5, cl = idx & 31;
            float xv = X[(size_t)(blockRow + r) * IN + c0 + cl];
            _Float16* arow = &sA[r][0];
            arow[cl] = (_Float16)siluf(xv);
            float bs[8];
            bspline8(xv, bs);
#pragma unroll
            for (int j = 0; j < 8; j++) arow[32 + cl * 8 + j] = (_Float16)bs[j];
        }
        __syncthreads();
        // Phase 2: 9 K-steps of 32 over this chunk
#pragma unroll
        for (int s = 0; s < 9; s++) {
            const int wk = (s == 0) ? c0 : (IN + 8 * c0 + (s - 1) * 32); // W column base
            HV bu[NT];
#pragma unroll
            for (int nt = 0; nt < NT; nt++) {
                const _Float16* wrow =
                    Wp + (size_t)(wave * (OUT / 8) + nt * 16 + ln) * K + wk + halfsel;
                bu[nt].h[0] = *(const v8h*)(wrow);
                bu[nt].h[1] = *(const v8h*)(wrow + 16);
            }
#pragma unroll
            for (int mt = 0; mt < 4; mt++) {
                const _Float16* arow = &sA[mt * 16 + ln][s * 32 + halfsel];
                HV au;
                au.h[0] = *(const v8h*)(arow);
                au.h[1] = *(const v8h*)(arow + 16);
#pragma unroll
                for (int nt = 0; nt < NT; nt++) {
                    acc[mt][nt] = __builtin_amdgcn_wmma_f32_16x16x32_f16(
                        false, au.v, false, bu[nt].v, (short)0, acc[mt][nt], false, false);
                }
            }
        }
    }
    // Epilogue: C/D layout -> row = base + (lane>=16?8:0) + r, col = lane&15
    const int rbase = (lane & 16) ? 8 : 0;
#pragma unroll
    for (int mt = 0; mt < 4; mt++)
#pragma unroll
        for (int nt = 0; nt < NT; nt++) {
            int col = wave * (OUT / 8) + nt * 16 + ln;
#pragma unroll
            for (int r = 0; r < 8; r++)
                Y[(size_t)(blockRow + mt * 16 + rbase + r) * OUT + col] = acc[mt][nt][r];
        }
}

__global__ __launch_bounds__(256) void layernorm256(float* __restrict__ h,
                                                    const float* __restrict__ g,
                                                    const float* __restrict__ b) {
    __shared__ float red[256];
    int row = blockIdx.x, t = threadIdx.x;
    float v = h[(size_t)row * 256 + t];
    red[t] = v;
    __syncthreads();
    for (int s = 128; s > 0; s >>= 1) { if (t < s) red[t] += red[t + s]; __syncthreads(); }
    float mu = red[0] * (1.0f / 256.0f);
    __syncthreads();
    float d = v - mu;
    red[t] = d * d;
    __syncthreads();
    for (int s = 128; s > 0; s >>= 1) { if (t < s) red[t] += red[t + s]; __syncthreads(); }
    float var = red[0] * (1.0f / 256.0f);
    h[(size_t)row * 256 + t] = d * rsqrtf(var + 1e-5f) * g[t] + b[t];
}

__global__ void zero_buf(float* p, int n) {
    int i = blockIdx.x * 256 + threadIdx.x;
    if (i < n) p[i] = 0.0f;
}

// LSTM gate update: one block per batch element b, one thread per channel c.
__global__ __launch_bounds__(256) void s2s_gates(const float* __restrict__ qstar,
                                                 const float* __restrict__ h_in,
                                                 float* __restrict__ h_out,
                                                 float* __restrict__ cst,
                                                 const float* __restrict__ w_ih,
                                                 const float* __restrict__ w_hh,
                                                 const float* __restrict__ b_ih,
                                                 const float* __restrict__ b_hh) {
    int b = blockIdx.x, c = threadIdx.x;
    int idx = b * 256 + c;
    float gv[4];
#pragma unroll
    for (int gg = 0; gg < 4; gg++) {
        int row = gg * 256 + c;
        float acc = b_ih[row] + b_hh[row];
        const float* wi = w_ih + (size_t)row * 512;
        const float* qs = qstar + b * 512;
        for (int k = 0; k < 512; k++) acc += wi[k] * qs[k];
        const float* wh = w_hh + (size_t)row * 256;
        const float* hh = h_in + b * 256;
        for (int k = 0; k < 256; k++) acc += wh[k] * hh[k];
        gv[gg] = acc;
    }
    float i_g = sigmf(gv[0]), f_g = sigmf(gv[1]), g_g = tanhf(gv[2]), o_g = sigmf(gv[3]);
    float cn  = f_g * cst[idx] + i_g * g_g;
    cst[idx]  = cn;
    h_out[idx] = o_g * tanhf(cn);
}

// e[n] = <hn[n], q[batch[n]]> ; per-block per-segment max partials (ordered-uint)
__global__ __launch_bounds__(256) void s2s_e_max(const float* __restrict__ hn,
                                                 const float* __restrict__ q,
                                                 const long long* __restrict__ batch,
                                                 float* __restrict__ e,
                                                 unsigned* __restrict__ maxp) {
    __shared__ float qs[16 * 256];
    __shared__ unsigned mx[16];
    int t = threadIdx.x;
    for (int i = t; i < 4096; i += 256) qs[i] = q[i];
    if (t < 16) mx[t] = 0u;
    __syncthreads();
    int n = blockIdx.x * 256 + t;
    int b = (int)batch[n];
    const float* hr = hn + (size_t)n * 256;
    const float* qr = qs + b * 256;
    float acc = 0.0f;
    for (int k = 0; k < 256; k++) acc += hr[k] * qr[k];
    e[n] = acc;
    atomicMax(&mx[b], f2ord(acc));
    __syncthreads();
    if (t < 16) maxp[blockIdx.x * 16 + t] = mx[t];
}

__global__ void s2s_max_reduce(const unsigned* __restrict__ maxp, float* __restrict__ emax) {
    int t = threadIdx.x;
    if (t < 16) {
        unsigned m = 0u;
        for (int blk = 0; blk < NB_ATT; blk++) {
            unsigned v = maxp[blk * 16 + t];
            m = (v > m) ? v : m;
        }
        emax[t] = ord2f(m);
    }
}

// ee = exp(e - emax[b]); per-block partial denom and weighted-sum r
__global__ __launch_bounds__(256) void s2s_soft(const float* __restrict__ hn,
                                                const long long* __restrict__ batch,
                                                const float* __restrict__ e,
                                                const float* __restrict__ emax,
                                                float* __restrict__ rpart,
                                                float* __restrict__ denp) {
    __shared__ float rsh[16 * 256];
    __shared__ float den[16];
    int t = threadIdx.x;
    for (int i = t; i < 4096; i += 256) rsh[i] = 0.0f;
    if (t < 16) den[t] = 0.0f;
    __syncthreads();
    int n = blockIdx.x * 256 + t;
    int b = (int)batch[n];
    float ee = __expf(e[n] - emax[b]);
    atomicAdd(&den[b], ee);
    const float* hr = hn + (size_t)n * 256;
    float* rb = rsh + b * 256;
    for (int k = 0; k < 256; k++) atomicAdd(&rb[k], ee * hr[k]);
    __syncthreads();
    for (int i = t; i < 4096; i += 256) rpart[(size_t)blockIdx.x * 4096 + i] = rsh[i];
    if (t < 16) denp[blockIdx.x * 16 + t] = den[t];
}

// fixed-order reduce of partials; build q_star = [q, r/denom]
__global__ void s2s_finish(const float* __restrict__ rpart, const float* __restrict__ denp,
                           const float* __restrict__ h_cur, float* __restrict__ qstar) {
    int b = blockIdx.x, c = threadIdx.x;
    float r = 0.0f, d = 0.0f;
    for (int blk = 0; blk < NB_ATT; blk++) {
        r += rpart[(size_t)blk * 4096 + b * 256 + c];
        d += denp[blk * 16 + b];
    }
    qstar[b * 512 + c]       = h_cur[b * 256 + c];
    qstar[b * 512 + 256 + c] = r / d;
}

// outputs: qstar[8192] | pos_z zeros[1536] | arange(512) | refl_z zeros[512]
__global__ void write_out(const float* __restrict__ qstar, float* __restrict__ out, int total) {
    int i = blockIdx.x * 256 + threadIdx.x;
    if (i >= total) return;
    float v;
    if (i < 8192)       v = qstar[i];
    else if (i < 9728)  v = 0.0f;
    else if (i < 10240) v = (float)(i - 9728);
    else                v = 0.0f;
    out[i] = v;
}

extern "C" void kernel_launch(void* const* d_in, const int* in_sizes, int n_in,
                              void* d_out, int out_size, void* d_ws, size_t ws_size,
                              hipStream_t stream) {
    (void)in_sizes; (void)n_in; (void)ws_size;
    const float*     x         = (const float*)d_in[0];
    const long long* batch     = (const long long*)d_in[3];
    const float*     base_w0   = (const float*)d_in[4];
    const float*     spline_w0 = (const float*)d_in[5];
    const float*     scaler0   = (const float*)d_in[6];
    const float*     base_w1   = (const float*)d_in[7];
    const float*     spline_w1 = (const float*)d_in[8];
    const float*     scaler1   = (const float*)d_in[9];
    const float*     ln_g      = (const float*)d_in[10];
    const float*     ln_b      = (const float*)d_in[11];
    const float*     w_ih      = (const float*)d_in[12];
    const float*     w_hh      = (const float*)d_in[13];
    const float*     b_ih      = (const float*)d_in[14];
    const float*     b_hh      = (const float*)d_in[15];

    char* ws = (char*)d_ws;
    size_t off = 0;
    auto carve = [&](size_t bytes) -> char* {
        char* p = ws + off;
        off = (off + bytes + 255) & ~(size_t)255;
        return p;
    };
    _Float16* Wp0   = (_Float16*)carve((size_t)128 * 576 * 2);
    _Float16* Wp1   = (_Float16*)carve((size_t)256 * 1152 * 2);
    float*    h1    = (float*)carve((size_t)NPTS * 128 * 4);
    float*    h2    = (float*)carve((size_t)NPTS * 256 * 4);
    float*    e     = (float*)carve((size_t)NPTS * 4);
    unsigned* maxp  = (unsigned*)carve((size_t)NB_ATT * 16 * 4);
    float*    denp  = (float*)carve((size_t)NB_ATT * 16 * 4);
    float*    rpart = (float*)carve((size_t)NB_ATT * 4096 * 4);
    float*    emax  = (float*)carve(16 * 4);
    float*    hA    = (float*)carve(4096 * 4);
    float*    hB    = (float*)carve(4096 * 4);
    float*    cst   = (float*)carve(4096 * 4);
    float*    qstar = (float*)carve(8192 * 4);

    // Pack augmented f16 weights (tiny, L2-resident)
    pack_w<64, 128><<<(128 * 576 + 255) / 256, 256, 0, stream>>>(base_w0, spline_w0, scaler0, Wp0);
    pack_w<128, 256><<<(256 * 1152 + 255) / 256, 256, 0, stream>>>(base_w1, spline_w1, scaler1, Wp1);

    // Fused KAN layers (WMMA)
    kan_gemm<64, 128><<<NPTS / 64, 256, 0, stream>>>(x, Wp0, h1);
    kan_gemm<128, 256><<<NPTS / 64, 256, 0, stream>>>(h1, Wp1, h2);

    // LayerNorm in place
    layernorm256<<<NPTS, 256, 0, stream>>>(h2, ln_g, ln_b);

    // Zero LSTM/attention state (hA, hB, cst, qstar are contiguous: 20480 floats)
    zero_buf<<<(20480 + 255) / 256, 256, 0, stream>>>(hA, 20480);

    // 8 Set2Set steps (h double-buffered)
    for (int step = 0; step < 8; step++) {
        float* h_in  = (step & 1) ? hB : hA;
        float* h_out = (step & 1) ? hA : hB;
        s2s_gates<<<16, 256, 0, stream>>>(qstar, h_in, h_out, cst, w_ih, w_hh, b_ih, b_hh);
        s2s_e_max<<<NB_ATT, 256, 0, stream>>>(h2, h_out, batch, e, maxp);
        s2s_max_reduce<<<1, 32, 0, stream>>>(maxp, emax);
        s2s_soft<<<NB_ATT, 256, 0, stream>>>(h2, batch, e, emax, rpart, denp);
        s2s_finish<<<16, 256, 0, stream>>>(rpart, denp, h_out, qstar);
    }

    write_out<<<(out_size + 255) / 256, 256, 0, stream>>>(qstar, (float*)d_out, out_size);
}
